// SoftSOMLayer_44341242364446
// MI455X (gfx1250) — matrix operand order repaired and checked by reading, not compiled, not run on hardware
//
#include <hip/hip_runtime.h>
#include <hip/hip_bf16.h>

typedef _Float16 v16h __attribute__((ext_vector_type(16)));
typedef _Float16 v8h  __attribute__((ext_vector_type(8)));
typedef float    v8f  __attribute__((ext_vector_type(8)));

#define NPROTO  1024          // 32*32 prototypes
#define DIM     512
#define BATCH   16384
#define MT      32            // batch rows per workgroup
#define TAU_INV 2.0f          // 1/0.5
#define EPS_R   1e-8f

// d_ws layout (bytes):
//  [0,       1048576)  _Float16 w16 [1024][512]   row-major f16 w   (GEMM1 B)
//  [1048576, 2097152)  _Float16 wT16[512][1024]   transposed f16 w  (GEMM2 B)
//  [2097152, 2101248)  float    ww  [1024]        ||w_n||^2
#define WS_W16   0
#define WS_WT16  1048576
#define WS_WW    2097152

// Main-kernel dynamic LDS layout (bytes):
//  [0,      131072)  float    llog[32][1024]  logits -> attention 'a' in-place
//  [131072, 163840)  _Float16 lx[32][512]     x tile, row-major f16
//  [163840, 167936)  float    wwl[1024]       ||w||^2 staged to LDS
//  [167936, 168064)  float    xx[32]          ||x||^2
#define SM_LLOG  0
#define SM_LX    131072
#define SM_WW    163840
#define SM_XX    167936
#define SM_BYTES 168064

// ---------------- prepass: w -> f16 (both layouts) + ||w||^2 ----------------
__global__ __launch_bounds__(256)
void softsom_prep(const float* __restrict__ w,
                  _Float16* __restrict__ w16,
                  _Float16* __restrict__ wT16,
                  float* __restrict__ ww) {
  const int n   = blockIdx.x;             // one prototype row per block
  const int tid = threadIdx.x;
  const int lane = tid & 31, wv = tid >> 5;
  const float* wr = w + (size_t)n * DIM;
  const float v0 = wr[tid];
  const float v1 = wr[tid + 256];
  w16[(size_t)n * DIM + tid]        = (_Float16)v0;
  w16[(size_t)n * DIM + tid + 256]  = (_Float16)v1;
  wT16[(size_t)tid * NPROTO + n]          = (_Float16)v0;
  wT16[(size_t)(tid + 256) * NPROTO + n]  = (_Float16)v1;
  float ss = v0 * v0 + v1 * v1;
#pragma unroll
  for (int o = 16; o >= 1; o >>= 1) ss += __shfl_xor(ss, o, 32);
  __shared__ float part[8];
  if (lane == 0) part[wv] = ss;
  __syncthreads();
  if (tid == 0) {
    float t = 0.f;
#pragma unroll
    for (int i = 0; i < 8; ++i) t += part[i];
    ww[n] = t;
  }
}

// ---------------- fused SoftSOM forward ----------------
__global__ __launch_bounds__(256)
void softsom_fused(const float* __restrict__ x,
                   const _Float16* __restrict__ w16,
                   const _Float16* __restrict__ wT16,
                   const float* __restrict__ wwg,
                   float* __restrict__ y) {
  extern __shared__ char smem[];
  float*    llog = (float*)(smem + SM_LLOG);
  _Float16* lx   = (_Float16*)(smem + SM_LX);
  float*    wwl  = (float*)(smem + SM_WW);
  float*    xx   = (float*)(smem + SM_XX);

  const int tid  = threadIdx.x;
  const int lane = tid & 31;              // wave32
  const int wv   = tid >> 5;              // 8 waves
  const int b0   = blockIdx.x * MT;

  const int mtile = wv & 1;               // b tile (0..1)
  const int grp   = wv >> 1;              // n-group (stage1) / d-group (stage3), 0..3
  const int offk  = (lane >> 4) * 8;      // ISA A/D layout: lane>=16 => +8 in K / M

  // ---- stage x tile (row-major f16, NT loads) + ||x||^2 ; stage ||w||^2 ----
  if (tid < 32) xx[tid] = 0.f;
#pragma unroll
  for (int i = 0; i < 4; ++i) wwl[tid + i * 256] = wwg[tid + i * 256];
  __syncthreads();
  {
    const int r  = tid >> 3;              // 0..31 batch row
    const int d0 = (tid & 7) * 64;        // 64 dims per thread
    const float* xp = x + (size_t)(b0 + r) * DIM + d0;
    _Float16* dst = lx + r * DIM + d0;
    float ss = 0.f;
#pragma unroll
    for (int i = 0; i < 64; ++i) {
      float v = __builtin_nontemporal_load(xp + i);   // x is single-touch: TH=NT
      ss += v * v;
      dst[i] = (_Float16)v;
    }
    atomicAdd(&xx[r], ss);                // ds_add_f32
  }
  __syncthreads();

  // ---- stage 1: logits = (2*x.wT - ||x||^2 - ||w||^2)/tau ----
  // D[M=b][N=n] = sum_k x[b][k] * wT[k][n].  A from LDS, B direct from L2-resident w16.
#pragma unroll 1
  for (int np = 0; np < 2; ++np) {
    const int nbase = grp * 256 + np * 128;
    v8f acc[8];
#pragma unroll
    for (int j = 0; j < 8; ++j)
#pragma unroll
      for (int i = 0; i < 8; ++i) acc[j][i] = 0.f;

#pragma unroll 1
    for (int kb = 0; kb < 16; ++kb) {
      const int k0 = kb * 32;
      // A fragment: x tile. lane: M=lane&15, K runs [k0+offk,+8) and [k0+16+offk,+8)
      const _Float16* ap = lx + (mtile * 16 + (lane & 15)) * DIM + k0 + offk;
      v8h alo = *(const v8h*)ap;
      v8h ahi = *(const v8h*)(ap + 16);
      v16h af = __builtin_shufflevector(alo, ahi,
                 0,1,2,3,4,5,6,7,8,9,10,11,12,13,14,15);
      // Preload ALL 8 B fragments (one big load clause), then 8 back-to-back WMMAs.
      const _Float16* bp = w16 + (size_t)(nbase + (lane & 15)) * DIM + k0 + offk;
      v16h bfv[8];
#pragma unroll
      for (int j = 0; j < 8; ++j) bfv[j] = __builtin_shufflevector(*(const v8h*)(bp + j * 16 * DIM), *(const v8h*)(bp + j * 16 * DIM + 16), 0,1,2,3,4,5,6,7,8,9,10,11,12,13,14,15);
      __builtin_prefetch(bp + 32, 0, 3);   // next K slab into WGP$
#pragma unroll
      for (int j = 0; j < 8; ++j) {
        acc[j] = __builtin_amdgcn_wmma_f32_16x16x32_f16(
                     false, af, false, bfv[j], (short)0, acc[j], false, false);
      }
    }
    // D layout: lane: N(n)=lane&15 ; M(b)=r + 8*(lane>=16)
    float xxv[8];
#pragma unroll
    for (int r = 0; r < 8; ++r) xxv[r] = xx[mtile * 16 + offk + r];
    const int nl = lane & 15;
#pragma unroll
    for (int j = 0; j < 8; ++j) {
      const int ncol = nbase + j * 16 + nl;
      const float wwv = wwl[ncol];
#pragma unroll
      for (int r = 0; r < 8; ++r) {
        const int brow = mtile * 16 + offk + r;
        llog[brow * NPROTO + ncol] = (2.f * acc[j][r] - xxv[r] - wwv) * TAU_INV;
      }
    }
  }
  __syncthreads();

  // ---- stage 2: per-axis softmax product + renorm (in-place, f32) ----
#pragma unroll 1
  for (int rr = 0; rr < 4; ++rr) {
    const int b = wv * 4 + rr;
    float* lp = llog + b * NPROTO + lane;
    float Lv[32];
#pragma unroll
    for (int gi = 0; gi < 32; ++gi) Lv[gi] = lp[gi * 32];
    float mr = Lv[0];
#pragma unroll
    for (int gi = 1; gi < 32; ++gi) mr = fmaxf(mr, Lv[gi]);
    float Sr = 0.f;
#pragma unroll
    for (int gi = 0; gi < 32; ++gi) Sr += __expf(Lv[gi] - mr);
    const float invSr = 1.f / Sr;
    float Av[32];
    float tot = 0.f;
#pragma unroll
    for (int gi = 0; gi < 32; ++gi) {
      const float v = Lv[gi];
      float mc = v;                       // softmax across gj: wave32 cross-lane
#pragma unroll
      for (int o = 16; o >= 1; o >>= 1) mc = fmaxf(mc, __shfl_xor(mc, o, 32));
      const float ec = __expf(v - mc);
      float Sc = ec;
#pragma unroll
      for (int o = 16; o >= 1; o >>= 1) Sc += __shfl_xor(Sc, o, 32);
      const float a = (__expf(v - mr) * invSr) * (ec / Sc);
      Av[gi] = a;
      tot += a;
    }
#pragma unroll
    for (int o = 16; o >= 1; o >>= 1) tot += __shfl_xor(tot, o, 32);
    const float invT = 1.f / (tot + EPS_R);
#pragma unroll
    for (int gi = 0; gi < 32; ++gi) lp[gi * 32] = Av[gi] * invT;
  }
  __syncthreads();

  // ---- stage 3: y = a @ w.  D[M=b][N=d], K=n.  B direct from L2-resident wT16. ----
  v8f acc3[8];
#pragma unroll
  for (int j = 0; j < 8; ++j)
#pragma unroll
    for (int i = 0; i < 8; ++i) acc3[j][i] = 0.f;

  const int bl3 = mtile * 16 + (lane & 15);
#pragma unroll 1
  for (int kb = 0; kb < 32; ++kb) {
    const int k0 = kb * 32;
    // A fragment from f32 attention in LDS, converted to f16 at load
    const float* ap = llog + bl3 * NPROTO + k0 + offk;
    v8f flo = *(const v8f*)ap;
    v8f fhi = *(const v8f*)(ap + 16);
    v16h af;
#pragma unroll
    for (int i = 0; i < 8; ++i) { af[i] = (_Float16)flo[i]; af[i + 8] = (_Float16)fhi[i]; }
    // Preload all 8 B fragments, then 8 back-to-back WMMAs.
    const _Float16* bp = wT16 + (size_t)(grp * 128 + (lane & 15)) * NPROTO + k0 + offk;
    v16h bfv[8];
#pragma unroll
    for (int j = 0; j < 8; ++j) bfv[j] = __builtin_shufflevector(*(const v8h*)(bp + j * 16 * NPROTO), *(const v8h*)(bp + j * 16 * NPROTO + 16), 0,1,2,3,4,5,6,7,8,9,10,11,12,13,14,15);
    __builtin_prefetch(bp + 32, 0, 3);        // next K slab into WGP$
#pragma unroll
    for (int j = 0; j < 8; ++j) {
      acc3[j] = __builtin_amdgcn_wmma_f32_16x16x32_f16(
                    false, af, false, bfv[j], (short)0, acc3[j], false, false);
    }
  }
  // store y (single-touch: TH=NT stores). lane: N(d)=lane&15 ; M(b)=r + 8*(lane>=16)
  const int dl   = lane & 15;
  const int brow = mtile * 16 + offk;
#pragma unroll
  for (int j = 0; j < 8; ++j) {
    const int dcol = grp * 128 + j * 16 + dl;
#pragma unroll
    for (int r = 0; r < 8; ++r) {
      __builtin_nontemporal_store(acc3[j][r],
          y + (size_t)(b0 + brow + r) * DIM + dcol);
    }
  }
}

extern "C" void kernel_launch(void* const* d_in, const int* in_sizes, int n_in,
                              void* d_out, int out_size, void* d_ws, size_t ws_size,
                              hipStream_t stream) {
  (void)in_sizes; (void)n_in; (void)out_size; (void)ws_size;
  const float* x = (const float*)d_in[0];            // [16384, 512] f32
  const float* w = (const float*)d_in[1];            // [32, 32, 512] f32 -> [1024, 512]
  float* y = (float*)d_out;                          // [16384, 512] f32

  _Float16* w16  = (_Float16*)((char*)d_ws + WS_W16);
  _Float16* wT16 = (_Float16*)((char*)d_ws + WS_WT16);
  float*    ww   = (float*)((char*)d_ws + WS_WW);

  hipLaunchKernelGGL(softsom_prep, dim3(NPROTO), dim3(256), 0, stream, w, w16, wT16, ww);
  hipLaunchKernelGGL(softsom_fused, dim3(BATCH / MT), dim3(256), SM_BYTES, stream,
                     x, w16, wT16, ww, y);
}
